// SSR_19275813225062
// MI455X (gfx1250) — matrix operand order repaired and checked by reading, not compiled, not run on hardware
//
#include <hip/hip_runtime.h>
#include <math.h>

// ---------------- problem constants ----------------
#define NB_CLS   5
#define NB_QRY   75
#define FEAT     640
#define NROW     80        // N = NB_CLS + NB_QRY
#define MIDX     81        // M = 1 + N
#define HID      4096
#define DNI_IN   465
#define DNI_PAD  480       // padded K for GEMM1 / padded R for GEMM3
#define LR_DNI   1e-3f

typedef __attribute__((ext_vector_type(16))) __bf16 v16bf;
typedef __attribute__((ext_vector_type(8)))  float  v8f;

// masked pairs of the 81x81 subgraph are exactly {(a,b): a<6, b>a}, 465 total,
// enumerated row-major (matches np.flatnonzero order).
__device__ __forceinline__ void jm_to_ab(int jm, int& a, int& b) {
    int off;
    if      (jm <  80) { a = 0; off =   0; }
    else if (jm < 159) { a = 1; off =  80; }
    else if (jm < 237) { a = 2; off = 159; }
    else if (jm < 314) { a = 3; off = 237; }
    else if (jm < 390) { a = 4; off = 314; }
    else               { a = 5; off = 390; }
    b = a + 1 + (jm - off);
}

// ---------------- prep: supp = mean(shots), feat = [supp ; query] ----------------
__global__ void k_prep(const float* __restrict__ fs, const float* __restrict__ fq,
                       float* __restrict__ feat) {
    int t = blockIdx.x * blockDim.x + threadIdx.x;
    if (t >= NROW * FEAT) return;
    int r = t / FEAT, f = t % FEAT;
    if (r < NB_CLS) {
        float s = 0.f;
        #pragma unroll
        for (int sh = 0; sh < 5; ++sh) s += fs[(r * 5 + sh) * FEAT + f];
        feat[t] = s * 0.2f;
    } else {
        feat[t] = fq[(r - NB_CLS) * FEAT + f];
    }
}

// ---------------- row L2-normalize ----------------
__global__ __launch_bounds__(256) void k_norm(const float* __restrict__ feat,
                                              float* __restrict__ fhat,
                                              float* __restrict__ invn) {
    int i = blockIdx.x;
    __shared__ float red[256];
    float s = 0.f;
    for (int f = threadIdx.x; f < FEAT; f += 256) {
        float v = feat[i * FEAT + f];
        s += v * v;
    }
    red[threadIdx.x] = s;
    __syncthreads();
    for (int st = 128; st; st >>= 1) {
        if (threadIdx.x < st) red[threadIdx.x] += red[threadIdx.x + st];
        __syncthreads();
    }
    float inv = 1.f / fmaxf(sqrtf(red[0]), 1e-12f);
    for (int f = threadIdx.x; f < FEAT; f += 256)
        fhat[i * FEAT + f] = feat[i * FEAT + f] * inv;
    if (threadIdx.x == 0) invn[i] = inv;
}

// ---------------- gram: sim = fhat @ fhat^T (80x80, tiny) ----------------
__global__ __launch_bounds__(128) void k_sim(const float* __restrict__ fhat,
                                             float* __restrict__ sim) {
    int i = blockIdx.x;
    __shared__ float rowi[FEAT];
    for (int f = threadIdx.x; f < FEAT; f += 128) rowi[f] = fhat[i * FEAT + f];
    __syncthreads();
    for (int j = threadIdx.x; j < NROW; j += 128) {
        float s = 0.f;
        for (int f = 0; f < FEAT; ++f) s += rowi[f] * fhat[j * FEAT + f];
        sim[i * NROW + j] = s;
    }
}

// ---------------- stable descending rank-argsort (== lax.top_k over full width) ----------------
__global__ __launch_bounds__(128) void k_sort(const float* __restrict__ sim,
                                              int* __restrict__ idx) {
    int i = blockIdx.x;
    __shared__ float row[NROW];
    for (int j = threadIdx.x; j < NROW; j += 128) row[j] = sim[i * NROW + j];
    __syncthreads();
    if (threadIdx.x == 0) idx[i * MIDX] = i;
    if (threadIdx.x < NB_CLS) {
        int j = threadIdx.x;
        float v = row[j];
        int r = 0;
        #pragma unroll
        for (int j2 = 0; j2 < NB_CLS; ++j2) {
            float v2 = row[j2];
            if (v2 > v || (v2 == v && j2 < j)) r++;
        }
        idx[i * MIDX + 1 + r] = j;
    }
    for (int j = threadIdx.x; j < NB_QRY; j += 128) {
        float v = row[NB_CLS + j];
        int r = 0;
        for (int j2 = 0; j2 < NB_QRY; ++j2) {
            float v2 = row[NB_CLS + j2];
            if (v2 > v || (v2 == v && j2 < j)) r++;
        }
        idx[i * MIDX + 1 + NB_CLS + r] = NB_CLS + j;
    }
}

// ---------------- gather masked subgraph -> bf16 activations Xb[80][DNI_PAD] ----------------
__global__ void k_gather(const float* __restrict__ sim, const int* __restrict__ idx,
                         __bf16* __restrict__ Xb) {
    int jm = blockIdx.x;      // 0..479
    int i  = threadIdx.x;     // 0..79
    float v = 0.f;
    if (jm < DNI_IN) {
        int a, b; jm_to_ab(jm, a, b);
        int p = idx[i * MIDX + a];
        int q = idx[i * MIDX + b];
        v = sim[p * NROW + q];
    }
    Xb[(size_t)i * DNI_PAD + jm] = (__bf16)v;
}

// ---------------- DNI GEMM: out[n][m] = sum_k W[n][k] * X[m][k]  (+bias, relu) ----------------
// 16 weight rows per workgroup, 8 waves split K (32-wide chunks), LDS reduce.
// Software-pipelined: next chunk's A + 5 B fragments load while current 5 WMMAs run.
// A-fragment (16x32 bf16): lane -> row = lane&15; K = lane<16 ? 0..15 : 16..31
// B-fragment (32x16 bf16): lane -> col = lane&15; K = lane<16 ? 0..15 : 16..31 (contig 32B load)
// Rows are CLAMPED (not masked): padded rows (GEMM3, 465..479) produce garbage that is
// never read. K tail (GEMM1, 465..479) uses a wave-uniform branch on the last chunk only;
// pad-K contributions are zero because Xb is zero-padded there.
__global__ __launch_bounds__(256) void dni_gemm(
        const float* __restrict__ W, const float* __restrict__ bias,
        const __bf16* __restrict__ Xb,
        int Rreal, int Kreal, int Kpad, int relu,
        __bf16* __restrict__ outB, int outStride,   // bf16 [80][outStride] (next B operand)
        float* __restrict__ outF)                   // f32  [Rpad][80]
{
    const int wave = threadIdx.x >> 5;
    const int lane = threadIdx.x & 31;
    const int l15  = lane & 15;
    const int hi   = lane >> 4;
    const int rowBase = blockIdx.x * 16;
    const int arow = min(rowBase + l15, Rreal - 1);     // clamp, never mask
    const float* __restrict__ pw = W + (size_t)arow * Kreal;

    // Per-lane base pointers for the 5 B fragments (col = t*16 + l15, K-half by `hi`)
    const __bf16* __restrict__ pb[5];
    #pragma unroll
    for (int t = 0; t < 5; ++t)
        pb[t] = Xb + (size_t)(t * 16 + l15) * Kpad + (hi ? 16 : 0);

    v8f acc[5];
    #pragma unroll
    for (int t = 0; t < 5; ++t) acc[t] = (v8f){0.f,0.f,0.f,0.f,0.f,0.f,0.f,0.f};

    auto loadA = [&](int kc) -> v16bf {
        v16bf a;
        const int k0 = kc + (hi ? 16 : 0);
        const int k1 = k0 + 8;
        if (kc + 32 <= Kreal) {                         // wave-uniform branch
            __builtin_prefetch(pw + kc + 512, 0, 0);    // 2 pipeline stages ahead
            const float4* p0 = (const float4*)(pw + k0);
            const float4* p1 = (const float4*)(pw + k1);
            float4 x0 = p0[0], x1 = p0[1], y0 = p1[0], y1 = p1[1];
            a[0]  = (__bf16)x0.x; a[1]  = (__bf16)x0.y;
            a[2]  = (__bf16)x0.z; a[3]  = (__bf16)x0.w;
            a[4]  = (__bf16)x1.x; a[5]  = (__bf16)x1.y;
            a[6]  = (__bf16)x1.z; a[7]  = (__bf16)x1.w;
            a[8]  = (__bf16)y0.x; a[9]  = (__bf16)y0.y;
            a[10] = (__bf16)y0.z; a[11] = (__bf16)y0.w;
            a[12] = (__bf16)y1.x; a[13] = (__bf16)y1.y;
            a[14] = (__bf16)y1.z; a[15] = (__bf16)y1.w;
        } else {                                        // K tail (last chunk of GEMM1 only)
            #pragma unroll
            for (int e = 0; e < 8; ++e) {
                int ka = k0 + e, kb = k1 + e;
                a[e]     = (__bf16)((ka < Kreal) ? pw[ka] : 0.f);
                a[8 + e] = (__bf16)((kb < Kreal) ? pw[kb] : 0.f);
            }
        }
        return a;
    };

    const int nch = Kpad >> 5;
    int c = wave;
    v16bf aN;
    v16bf bN[5];
    if (c < nch) {
        aN = loadA(c << 5);
        #pragma unroll
        for (int t = 0; t < 5; ++t) bN[t] = *(const v16bf*)(pb[t] + (c << 5));
    }
    while (c < nch) {
        const v16bf aC = aN;
        v16bf bC[5];
        #pragma unroll
        for (int t = 0; t < 5; ++t) bC[t] = bN[t];

        const int cn = c + 8;
        if (cn < nch) {                                 // prefetch next pipeline stage
            aN = loadA(cn << 5);
            #pragma unroll
            for (int t = 0; t < 5; ++t) bN[t] = *(const v16bf*)(pb[t] + (cn << 5));
        }
        #pragma unroll
        for (int t = 0; t < 5; ++t)
            acc[t] = __builtin_amdgcn_wmma_f32_16x16x32_bf16(
                         false, aC, false, bC[t], (short)0, acc[t],
                         false, false);
        c = cn;
    }

    // cross-wave K reduction + epilogue
    __shared__ float red[8][16][80];           // 40 KB
    #pragma unroll
    for (int t = 0; t < 5; ++t)
        #pragma unroll
        for (int v = 0; v < 8; ++v)
            red[wave][v + 8 * hi][t * 16 + l15] = acc[t][v];
    __syncthreads();

    for (int p = threadIdx.x; p < 16 * 80; p += 256) {
        int nl = p / 80, m = p % 80;
        float s = 0.f;
        #pragma unroll
        for (int wv = 0; wv < 8; ++wv) s += red[wv][nl][m];
        int n = rowBase + nl;
        if (bias && n < Rreal) s += bias[n];
        if (relu) s = fmaxf(s, 0.f);
        if (outB) outB[(size_t)m * outStride + n] = (__bf16)s;
        if (outF) outF[(size_t)n * 80 + m] = s;
    }
}

// ---------------- zero / scatter cotangent into d_sim ----------------
__global__ void k_zero(float* __restrict__ p, int n) {
    int t = blockIdx.x * blockDim.x + threadIdx.x;
    if (t < n) p[t] = 0.f;
}

__global__ void k_scatter(const float* __restrict__ yt, const int* __restrict__ idx,
                          float* __restrict__ dsim) {
    int jm = blockIdx.x;      // 0..464
    int i  = threadIdx.x;     // 0..79
    int a, b; jm_to_ab(jm, a, b);
    int p = idx[i * MIDX + a];
    int q = idx[i * MIDX + b];
    atomicAdd(&dsim[p * NROW + q], yt[(size_t)jm * 80 + i]);
}

// ---------------- backward through gram+norm, SGD update on selected rows ----------------
__global__ __launch_bounds__(256) void k_backupd(const float* __restrict__ dsim,
                                                 const float* __restrict__ fhat,
                                                 const float* __restrict__ invn,
                                                 float* __restrict__ feat,
                                                 int rowOff) {
    int i = blockIdx.x + rowOff;
    __shared__ float g[NROW];
    __shared__ float df[FEAT];
    __shared__ float red[256];
    for (int q = threadIdx.x; q < NROW; q += 256)
        g[q] = dsim[i * NROW + q] + dsim[q * NROW + i];   // (G + G^T) row
    __syncthreads();
    float dot = 0.f;
    for (int f = threadIdx.x; f < FEAT; f += 256) {
        float s = 0.f;
        for (int q = 0; q < NROW; ++q) s += g[q] * fhat[q * FEAT + f];
        df[f] = s;
        dot += s * fhat[i * FEAT + f];
    }
    red[threadIdx.x] = dot;
    __syncthreads();
    for (int st = 128; st; st >>= 1) {
        if (threadIdx.x < st) red[threadIdx.x] += red[threadIdx.x + st];
        __syncthreads();
    }
    dot = red[0];
    float inv = invn[i];
    for (int f = threadIdx.x; f < FEAT; f += 256) {
        float d = (df[f] - fhat[i * FEAT + f] * dot) * inv;
        feat[i * FEAT + f] -= LR_DNI * d;
    }
}

// ---------------- final: out[q][c] = scale * <fhat_c, fhat_{5+q}> ----------------
__global__ void k_final(const float* __restrict__ fhat, const float* __restrict__ scale,
                        float* __restrict__ out) {
    int t = blockIdx.x * blockDim.x + threadIdx.x;
    if (t >= NB_QRY * NB_CLS) return;
    int qy = t / NB_CLS, c = t % NB_CLS;
    float s = 0.f;
    for (int f = 0; f < FEAT; ++f)
        s += fhat[c * FEAT + f] * fhat[(NB_CLS + qy) * FEAT + f];
    out[t] = s * scale[0];
}

// ---------------- orchestration ----------------
extern "C" void kernel_launch(void* const* d_in, const int* in_sizes, int n_in,
                              void* d_out, int out_size, void* d_ws, size_t ws_size,
                              hipStream_t stream) {
    (void)in_sizes; (void)n_in; (void)out_size; (void)ws_size;

    const float* fs    = (const float*)d_in[0];
    const float* fq    = (const float*)d_in[1];
    const float* scale = (const float*)d_in[2];
    const float* Wset[2][6];
    for (int s = 0; s < 2; ++s)
        for (int j = 0; j < 6; ++j)
            Wset[s][j] = (const float*)d_in[3 + s * 6 + j];

    // scratch carve (256B aligned); total ~2.0 MB
    char* base = (char*)d_ws;
    size_t off = 0;
    auto carve = [&](size_t bytes) -> void* {
        off = (off + 255) & ~(size_t)255;
        void* p = base + off;
        off += bytes;
        return p;
    };
    float*  feat = (float*)carve(NROW * FEAT * 4);
    float*  fhat = (float*)carve(NROW * FEAT * 4);
    float*  invn = (float*)carve(NROW * 4);
    float*  sim  = (float*)carve(NROW * NROW * 4);
    int*    idx  = (int*)  carve(NROW * MIDX * 4);
    __bf16* Xb   = (__bf16*)carve((size_t)NROW * DNI_PAD * 2);
    __bf16* h1b  = (__bf16*)carve((size_t)NROW * HID * 2);
    __bf16* h2b  = (__bf16*)carve((size_t)NROW * HID * 2);
    float*  yt   = (float*)carve((size_t)DNI_PAD * 80 * 4);
    float*  dsim = (float*)carve(NROW * NROW * 4);

    k_prep<<<(NROW * FEAT + 255) / 256, 256, 0, stream>>>(fs, fq, feat);

    for (int step = 0; step < 3; ++step) {
        for (int m = 0; m < 2; ++m) {           // m=0: supp update (ds), m=1: query update (dq)
            const float* const* Wp = Wset[m];
            k_norm<<<NROW, 256, 0, stream>>>(feat, fhat, invn);
            k_sim<<<NROW, 128, 0, stream>>>(fhat, sim);
            k_sort<<<NROW, 128, 0, stream>>>(sim, idx);
            k_gather<<<DNI_PAD, 80, 0, stream>>>(sim, idx, Xb);
            // h1 = relu(x @ W1^T + b1)
            dni_gemm<<<HID / 16, 256, 0, stream>>>(Wp[0], Wp[1], Xb,
                                                   HID, DNI_IN, DNI_PAD, 1,
                                                   h1b, HID, nullptr);
            // h2 = relu(h1 @ W2^T + b2)
            dni_gemm<<<HID / 16, 256, 0, stream>>>(Wp[2], Wp[3], h1b,
                                                   HID, HID, HID, 1,
                                                   h2b, HID, nullptr);
            // y = h2 @ W3^T + b3
            dni_gemm<<<DNI_PAD / 16, 256, 0, stream>>>(Wp[4], Wp[5], h2b,
                                                       DNI_IN, HID, HID, 0,
                                                       nullptr, 0, yt);
            k_zero<<<(NROW * NROW + 255) / 256, 256, 0, stream>>>(dsim, NROW * NROW);
            k_scatter<<<DNI_IN, 80, 0, stream>>>(yt, idx, dsim);
            k_backupd<<<m ? NB_QRY : NB_CLS, 256, 0, stream>>>(dsim, fhat, invn, feat,
                                                               m ? NB_CLS : 0);
        }
    }

    k_norm<<<NROW, 256, 0, stream>>>(feat, fhat, invn);
    k_final<<<(NB_QRY * NB_CLS + 255) / 256, 256, 0, stream>>>(fhat, scale, (float*)d_out);
}
